// MultiHeadAttention_67654324847333
// MI455X (gfx1250) — compile-verified, matrix operand order checked by reading, not run on hardware
//
#include <hip/hip_runtime.h>
#include <hip/hip_bf16.h>

// ---------------------------------------------------------------------------
// MHA forward for MI455X (gfx1250): bf16 WMMA + TDM (tensor_load_to_lds)
// staging of shared A-tiles, fp32 accumulate.
// ---------------------------------------------------------------------------

typedef __attribute__((ext_vector_type(16))) __bf16 v16bf;
typedef __attribute__((ext_vector_type(8)))  __bf16 v8bf;
typedef __attribute__((ext_vector_type(8)))  float  v8f;
typedef __attribute__((ext_vector_type(4)))  unsigned int u32x4;
typedef __attribute__((ext_vector_type(8)))  int i32x8;
typedef __attribute__((ext_vector_type(4)))  int i32x4;

#define HIDDEN 2048
#define NHEADS 16
#define HDIM   128
#define BATCH  2
#define SEQ    2048
#define MROWS  (BATCH * SEQ)            // 4096
#define ATTN_SCALE 0.08838834764831845f // 1/sqrt(128)

#define KCHUNK 128                      // k-elements staged per TDM transfer
#define NCHUNK (HIDDEN / KCHUNK)        // 16
#define LDSP   (KCHUNK + 8)             // LDS row pitch (+16B TDM pad -> no bank conflicts)

// ---- fragment helpers ------------------------------------------------------
// 16-bit A-fragment (16x32, MxK) per CDNA5 ISA layout:
//   lane l: m = l&15, h = l>>4
//   e[0..7]  = A[m, kc + 8h + 0..7]
//   e[8..15] = A[m, kc + 16 + 8h + 0..7]
// B operand (32x16, KxN) mirrored from column-major source (row-major W/K/Vt).
static __device__ inline v16bf load_frag(const __bf16* __restrict__ base, int ld,
                                         int row, int kc, int lane) {
  const int m = lane & 15;
  const int h = lane >> 4;
  const __bf16* p = base + (size_t)(row + m) * (size_t)ld + kc + 8 * h;
  union { v16bf v; v8bf h2[2]; } u;
  u.h2[0] = *reinterpret_cast<const v8bf*>(p);
  u.h2[1] = *reinterpret_cast<const v8bf*>(p + 16);
  return u.v;
}

static __device__ inline v16bf load_frag_lds(const __bf16* base, int pitch,
                                             int row, int kc, int lane) {
  const int m = lane & 15;
  const int h = lane >> 4;
  const __bf16* p = base + (row + m) * pitch + kc + 8 * h;
  union { v16bf v; v8bf h2[2]; } u;
  u.h2[0] = *reinterpret_cast<const v8bf*>(p);
  u.h2[1] = *reinterpret_cast<const v8bf*>(p + 16);
  return u.v;
}

static __device__ inline v8f wmma_bf16(v16bf a, v16bf b, v8f c) {
  return __builtin_amdgcn_wmma_f32_16x16x32_bf16(
      false, a, false, b, (short)0, c, false, false);
}

// ---- TDM: move a 32-row x 128-col bf16 tile of a [4096 x 2048] matrix ------
// D# per CDNA5 ISA. Row padding: every 256B (one tile
// row) insert 16B -> LDS pitch 136 elements.
// NOTE: this toolchain (clang-23/therock) uses the 6-arg builtin form:
//   (u32x4 g0, i32x8 g1, i32x4 g2, i32x4 g3, i32x8 extra, i32 cpol)
static __device__ inline void tdm_load_tile_a(unsigned long long gaddr, unsigned int laddr) {
  u32x4 g0 = { 1u,                                   // count=1, user mode
               laddr,                                // lds_addr (bytes)
               (unsigned int)gaddr,                  // global_addr[31:0]
               (unsigned int)(gaddr >> 32) | (2u << 30) }; // addr[56:32] | type=2
  i32x8 g1 = { (int)((1u << 16)        // data_size = 2 bytes
                   | (1u << 20)        // pad_enable
                   | (5u << 22)        // pad_interval: 64 DWORDs = 256B (one row)
                   | (3u << 25)),      // pad_amount:   4 DWORDs = 16B
               (int)(2048u << 16),     // tensor_dim0 = 2048 (low16 at bits 63:48)
               (int)(4096u << 16),     // dim0 hi16=0 | tensor_dim1 = 4096 (low16)
               (int)(128u << 16),      // dim1 hi16=0 | tile_dim0 = 128
               32,                     // tile_dim1 = 32 | tile_dim2 = 0
               2048,                   // tensor_dim0_stride[31:0] = 2048
               0, 0 };                 // stride0 hi | tensor_dim1_stride = 0
  i32x4 gz4 = { 0, 0, 0, 0 };
  i32x8 gz8 = { 0, 0, 0, 0, 0, 0, 0, 0 };
  __builtin_amdgcn_tensor_load_to_lds(g0, g1, gz4, gz4, gz8, 0);
}

// ---- shared GEMM core: C(32x32 per wave) = X[m0..] @ W[n0..]^T -------------
// A-tile (32 x KCHUNK) staged in LDS by TDM, double buffered; B streamed.
static __device__ inline void gemm_core(const __bf16* __restrict__ X,
                                        const __bf16* __restrict__ W,
                                        __bf16* tileA, int m0, int n0,
                                        int tid, v8f (&c)[2][2]) {
  const int lane = tid & 31;
  const int wid  = tid >> 5;
  const unsigned long long xrow =
      (unsigned long long)(uintptr_t)(X + (size_t)m0 * HIDDEN);
  const unsigned int lds0  = (unsigned int)(uintptr_t)tileA; // flat low32 = LDS offset
  const unsigned int bufsz = 32u * LDSP * 2u;

  if (wid == 0) tdm_load_tile_a(xrow, lds0);

  for (int ch = 0; ch < NCHUNK; ++ch) {
    if (wid == 0) {
      if (ch + 1 < NCHUNK) {
        tdm_load_tile_a(xrow + (unsigned long long)(ch + 1) * (KCHUNK * 2),
                        lds0 + (unsigned int)((ch + 1) & 1) * bufsz);
        __builtin_amdgcn_s_wait_tensorcnt(1);  // chunk ch landed (in-order)
      } else {
        __builtin_amdgcn_s_wait_tensorcnt(0);
      }
    }
    __syncthreads();  // buf[ch&1] ready for all waves

    const __bf16* A = tileA + (ch & 1) * (32 * LDSP);
    const int kbase = ch * KCHUNK;
#pragma unroll
    for (int kk = 0; kk < KCHUNK; kk += 32) {
      v16bf a0 = load_frag_lds(A, LDSP, 0,  kk, lane);
      v16bf a1 = load_frag_lds(A, LDSP, 16, kk, lane);
      v16bf b0 = load_frag(W, HIDDEN, n0,      kbase + kk, lane);
      v16bf b1 = load_frag(W, HIDDEN, n0 + 16, kbase + kk, lane);
      c[0][0] = wmma_bf16(a0, b0, c[0][0]);
      c[0][1] = wmma_bf16(a0, b1, c[0][1]);
      c[1][0] = wmma_bf16(a1, b0, c[1][0]);
      c[1][1] = wmma_bf16(a1, b1, c[1][1]);
    }
    __syncthreads();  // done with buf[ch&1] before TDM recycles it
  }
}

// ---- kernel 1: fp32 -> bf16 ------------------------------------------------
__global__ __launch_bounds__(256) void cvt_bf16_kernel(const float* __restrict__ in,
                                                       __bf16* __restrict__ out, int n) {
  for (int i = blockIdx.x * blockDim.x + threadIdx.x; i < n; i += gridDim.x * blockDim.x)
    out[i] = (__bf16)in[i];
}

// ---- kernel 2: fused QKV projection ---------------------------------------
__global__ __launch_bounds__(256) void qkv_kernel(
    const __bf16* __restrict__ X,
    const __bf16* __restrict__ Wq, const __bf16* __restrict__ Wk, const __bf16* __restrict__ Wv,
    const float* __restrict__ bq, const float* __restrict__ bk, const float* __restrict__ bv,
    __bf16* __restrict__ Q, __bf16* __restrict__ K, __bf16* __restrict__ Vt) {
  __shared__ __align__(16) __bf16 tileA[2 * 32 * LDSP];
  const int lane = threadIdx.x & 31;
  const int wid  = threadIdx.x >> 5;
  const int hh   = lane >> 4;
  const int ln   = lane & 15;
  const int n0 = blockIdx.x * 256 + wid * 32;
  const int m0 = blockIdx.y * 32;
  const int z  = blockIdx.z;
  const __bf16* W   = (z == 0) ? Wq : (z == 1) ? Wk : Wv;
  const float* bias = (z == 0) ? bq : (z == 1) ? bk : bv;

  v8f c[2][2] = {};
  gemm_core(X, W, tileA, m0, n0, threadIdx.x, c);

#pragma unroll
  for (int mt = 0; mt < 2; ++mt) {
#pragma unroll
    for (int nt = 0; nt < 2; ++nt) {
      const int col  = n0 + nt * 16 + ln;      // hidden column -> (head, d)
      const float bv_ = bias[col];
      const int head = col >> 7;
      const int d    = col & 127;
      const int row0 = m0 + mt * 16 + 8 * hh;  // 8 consecutive rows per lane
      const int bb   = row0 >> 11;
      const int s0   = row0 & 2047;
      if (z == 2) {
        // V transposed: Vt[b, head, d, s] -> 8 consecutive s => one 16B store
        v8bf pk;
#pragma unroll
        for (int i = 0; i < 8; ++i) pk[i] = (__bf16)(c[mt][nt][i] + bv_);
        *reinterpret_cast<v8bf*>(
            Vt + ((size_t)((bb * NHEADS + head) * HDIM + d)) * SEQ + s0) = pk;
      } else {
        __bf16* dst = (z == 0) ? Q : K;
#pragma unroll
        for (int i = 0; i < 8; ++i)
          dst[((size_t)((bb * NHEADS + head) * SEQ) + (s0 + i)) * HDIM + d] =
              (__bf16)(c[mt][nt][i] + bv_);
      }
    }
  }
}

// ---- kernel 3: flash attention --------------------------------------------
__global__ __launch_bounds__(128) void attn_kernel(
    const __bf16* __restrict__ Q, const __bf16* __restrict__ K,
    const __bf16* __restrict__ Vt, const float* __restrict__ mask,
    __bf16* __restrict__ ctx) {
  __shared__ __bf16 plds[4 * 16 * 32];  // per-wave 16x32 P tile
  const int lane = threadIdx.x & 31;
  const int wid  = threadIdx.x >> 5;
  const int hh   = lane >> 4;
  const int ln   = lane & 15;
  const int b  = blockIdx.z;
  const int hd = blockIdx.y;
  const int qbase = (blockIdx.x * 4 + wid) * 16;

  const __bf16* Qp = Q  + (size_t)(b * NHEADS + hd) * SEQ * HDIM;
  const __bf16* Kp = K  + (size_t)(b * NHEADS + hd) * SEQ * HDIM;
  const __bf16* Vp = Vt + (size_t)(b * NHEADS + hd) * HDIM * SEQ;
  const float*  mp = mask + (size_t)b * SEQ;
  __bf16* myp = plds + wid * (16 * 32);

  v16bf qf[4];
#pragma unroll
  for (int c = 0; c < 4; ++c) qf[c] = load_frag(Qp, HDIM, qbase, c * 32, lane);

  v8f acc[8] = {};
  float mrow[8], lrow[8];
#pragma unroll
  for (int i = 0; i < 8; ++i) { mrow[i] = -1e30f; lrow[i] = 0.0f; }

  for (int kt = 0; kt < SEQ; kt += 32) {
    v8f s0 = {}, s1 = {};
#pragma unroll
    for (int c = 0; c < 4; ++c) {
      v16bf kf = load_frag(Kp, HDIM, kt, c * 32, lane);
      s0 = wmma_bf16(qf[c], kf, s0);
    }
#pragma unroll
    for (int c = 0; c < 4; ++c) {
      v16bf kf = load_frag(Kp, HDIM, kt + 16, c * 32, lane);
      s1 = wmma_bf16(qf[c], kf, s1);
    }
    const float mk0 = mp[kt + ln];
    const float mk1 = mp[kt + 16 + ln];

    float p0a[8], p1a[8];
#pragma unroll
    for (int i = 0; i < 8; ++i) {
      float a  = s0[i] * ATTN_SCALE + mk0;
      float b2 = s1[i] * ATTN_SCALE + mk1;
      float r = fmaxf(a, b2);
#pragma unroll
      for (int off = 1; off < 16; off <<= 1) r = fmaxf(r, __shfl_xor(r, off, 32));
      const float mnew  = fmaxf(mrow[i], r);
      const float alpha = __expf(mrow[i] - mnew);
      mrow[i] = mnew;
      const float p0 = __expf(a - mnew);
      const float p1 = __expf(b2 - mnew);
      p0a[i] = p0; p1a[i] = p1;
      float rs = p0 + p1;
#pragma unroll
      for (int off = 1; off < 16; off <<= 1) rs += __shfl_xor(rs, off, 32);
      lrow[i] = lrow[i] * alpha + rs;
#pragma unroll
      for (int t = 0; t < 8; ++t) acc[t][i] *= alpha;
    }

    // C-layout -> A-fragment via per-wave LDS round trip
#pragma unroll
    for (int i = 0; i < 8; ++i) {
      const int row = i + 8 * hh;
      myp[row * 32 + ln]      = (__bf16)p0a[i];
      myp[row * 32 + 16 + ln] = (__bf16)p1a[i];
    }
    asm volatile("s_wait_dscnt 0" ::: "memory");  // intra-wave LDS RAW
    v16bf pf;
    {
      union { v16bf v; v8bf h2[2]; } u;
      u.h2[0] = *reinterpret_cast<const v8bf*>(myp + ln * 32 + 8 * hh);
      u.h2[1] = *reinterpret_cast<const v8bf*>(myp + ln * 32 + 16 + 8 * hh);
      pf = u.v;
    }

#pragma unroll
    for (int t = 0; t < 8; ++t) {
      v16bf vf = load_frag(Vp, SEQ, t * 16, kt, lane);
      acc[t] = wmma_bf16(pf, vf, acc[t]);
    }
  }

#pragma unroll
  for (int t = 0; t < 8; ++t) {
#pragma unroll
    for (int i = 0; i < 8; ++i) {
      const float v  = acc[t][i] / lrow[i];
      const int s    = qbase + i + 8 * hh;
      const int col  = hd * HDIM + t * 16 + ln;
      ctx[(size_t)(b * SEQ + s) * HIDDEN + col] = (__bf16)v;
    }
  }
}

// ---- kernel 4: output projection (fp32 out) --------------------------------
__global__ __launch_bounds__(256) void out_gemm_kernel(
    const __bf16* __restrict__ X, const __bf16* __restrict__ Wo,
    const float* __restrict__ bo, float* __restrict__ out) {
  __shared__ __align__(16) __bf16 tileA[2 * 32 * LDSP];
  const int lane = threadIdx.x & 31;
  const int wid  = threadIdx.x >> 5;
  const int hh   = lane >> 4;
  const int ln   = lane & 15;
  const int n0 = blockIdx.x * 256 + wid * 32;
  const int m0 = blockIdx.y * 32;

  v8f c[2][2] = {};
  gemm_core(X, Wo, tileA, m0, n0, threadIdx.x, c);

#pragma unroll
  for (int mt = 0; mt < 2; ++mt) {
#pragma unroll
    for (int nt = 0; nt < 2; ++nt) {
      const int col = n0 + nt * 16 + ln;
      const float bv_ = bo[col];
#pragma unroll
      for (int i = 0; i < 8; ++i) {
        const int row = m0 + mt * 16 + 8 * hh + i;
        out[(size_t)row * HIDDEN + col] = c[mt][nt][i] + bv_;
      }
    }
  }
}

// ---- launch ----------------------------------------------------------------
extern "C" void kernel_launch(void* const* d_in, const int* in_sizes, int n_in,
                              void* d_out, int out_size, void* d_ws, size_t ws_size,
                              hipStream_t stream) {
  const float* hs   = (const float*)d_in[0];
  const float* mask = (const float*)d_in[1];
  const float* Wq   = (const float*)d_in[2];
  const float* bq   = (const float*)d_in[3];
  const float* Wk   = (const float*)d_in[4];
  const float* bk   = (const float*)d_in[5];
  const float* Wv   = (const float*)d_in[6];
  const float* bv   = (const float*)d_in[7];
  const float* Wo   = (const float*)d_in[8];
  const float* bo   = (const float*)d_in[9];

  const size_t NHS = (size_t)MROWS * HIDDEN;   // 8388608
  const size_t NW  = (size_t)HIDDEN * HIDDEN;  // 4194304

  __bf16* ws   = (__bf16*)d_ws;
  __bf16* hsb  = ws;
  __bf16* wqb  = hsb + NHS;
  __bf16* wkb  = wqb + NW;
  __bf16* wvb  = wkb + NW;
  __bf16* wob  = wvb + NW;
  __bf16* Qb   = wob + NW;
  __bf16* Kb   = Qb + NHS;
  __bf16* Vtb  = Kb + NHS;
  __bf16* ctxb = Vtb + NHS;

  cvt_bf16_kernel<<<4096, 256, 0, stream>>>(hs, hsb, (int)NHS);
  cvt_bf16_kernel<<<4096, 256, 0, stream>>>(Wq, wqb, (int)NW);
  cvt_bf16_kernel<<<4096, 256, 0, stream>>>(Wk, wkb, (int)NW);
  cvt_bf16_kernel<<<4096, 256, 0, stream>>>(Wv, wvb, (int)NW);
  cvt_bf16_kernel<<<4096, 256, 0, stream>>>(Wo, wob, (int)NW);

  qkv_kernel<<<dim3(HIDDEN / 256, MROWS / 32, 3), 256, 0, stream>>>(
      hsb, wqb, wkb, wvb, bq, bk, bv, Qb, Kb, Vtb);

  attn_kernel<<<dim3(SEQ / 64, NHEADS, BATCH), 128, 0, stream>>>(
      Qb, Kb, Vtb, mask, ctxb);

  out_gemm_kernel<<<dim3(HIDDEN / 256, MROWS / 32), 256, 0, stream>>>(
      ctxb, wob, bo, (float*)d_out);
}
